// Head_32006096289987
// MI455X (gfx1250) — matrix operand order repaired and checked by reading, not compiled, not run on hardware
//
#include <hip/hip_runtime.h>
#include <hip/hip_bf16.h>

typedef _Float16 v16h __attribute__((ext_vector_type(16)));
typedef float    v8f  __attribute__((ext_vector_type(8)));

typedef unsigned int tdm_u32x4 __attribute__((ext_vector_type(4)));
typedef int          tdm_i32x4 __attribute__((ext_vector_type(4)));
typedef int          tdm_i32x8 __attribute__((ext_vector_type(8)));

union F16Frag { v16h v; _Float16 h[16]; unsigned u[8]; };
union F32Frag { v8f  v; float f[8]; };

#define BB 4
#define SS 4096
#define DM 768
#define DH 64

// A-fragment K mapping (16-bit A 16x32, wave32), per ISA 7.12.2:
// lanes 0-15: VGPR0..3 -> K=0..7, VGPR4..7 -> K=16..23; lanes 16-31: +8
__device__ __forceinline__ int kmapA(int v, int hi) {
    return 2 * (v & 3) + 16 * (v >> 2) + 8 * hi;
}
// B-fragment K mapping (16-bit B 32x16, wave32), same as A:
// lanes 0-15: VGPR0..3 -> K=0..7, VGPR4..7 -> K=16..23; lanes 16-31: +8
__device__ __forceinline__ int kmapB(int v, int hi) {
    return 2 * (v & 3) + 16 * (v >> 2) + 8 * hi;
}

__device__ __forceinline__ v8f wmma16(v16h a, v16h b, v8f c) {
    return __builtin_amdgcn_wmma_f32_16x16x32_f16(
        /*neg_a=*/false, a, /*neg_b=*/false, b,
        /*c_mod=*/(short)0, c, /*reuse_a=*/false, /*reuse_b=*/false);
}

__device__ __forceinline__ float red_max16(float t) {
    #pragma unroll
    for (int m = 1; m < 16; m <<= 1) t = fmaxf(t, __shfl_xor(t, m, 32));
    return t;
}
__device__ __forceinline__ float red_sum16(float t) {
    #pragma unroll
    for (int m = 1; m < 16; m <<= 1) t += __shfl_xor(t, m, 32);
    return t;
}

// ---------------------------------------------------------------------------
// TDM: DMA a 2-D tile (rows x 64 f16, row stride 64) from global into LDS.
// D# group0: count=1 | lds_addr | global_addr(57b) | type=2.
// D# group1: data_size=2B, tensor_dim0/1, tile_dim0/1, dim0_stride.
// ---------------------------------------------------------------------------
__device__ __forceinline__ void tdm_load_tile(
    unsigned lds_byte_addr, const void* gptr,
    unsigned tile_d0, unsigned tile_d1,
    unsigned tensor_d0, unsigned tensor_d1, unsigned stride0)
{
    unsigned long long ga = (unsigned long long)gptr;
    tdm_u32x4 g0;
    g0[0] = 1u;                                        // count=1, user mode
    g0[1] = lds_byte_addr;                             // LDS address (bytes)
    g0[2] = (unsigned)(ga & 0xffffffffull);            // global_addr[31:0]
    g0[3] = (unsigned)((ga >> 32) & 0x01ffffffull)     // global_addr[56:32]
          | (2u << 30);                                // type=2 ("image")
    tdm_i32x8 g1;
    g1[0] = (int)(1u << 16);                           // data_size=1 -> 2 bytes
    g1[1] = (int)((tensor_d0 & 0xffffu) << 16);        // tensor_dim0[15:0]
    g1[2] = (int)(((tensor_d0 >> 16) & 0xffffu)
          | ((tensor_d1 & 0xffffu) << 16));            // dim0 hi | dim1 lo
    g1[3] = (int)(((tensor_d1 >> 16) & 0xffffu)
          | ((tile_d0 & 0xffffu) << 16));              // dim1 hi | tile_dim0
    g1[4] = (int)(tile_d1 & 0xffffu);                  // tile_dim1 (tile_dim2=0)
    g1[5] = (int)stride0;                              // tensor_dim0_stride lo
    g1[6] = 0;                                         // stride0 hi | stride1 lo
    g1[7] = 0;
    tdm_i32x4 z4 = {0, 0, 0, 0};
#if defined(__clang_major__) && (__clang_major__ >= 23)
    tdm_i32x8 z8 = {0, 0, 0, 0, 0, 0, 0, 0};
    __builtin_amdgcn_tensor_load_to_lds(g0, g1, z4, z4, z8, 0);
#else
    __builtin_amdgcn_tensor_load_to_lds(g0, g1, z4, z4, 0);
#endif
}

// ---------------------------------------------------------------------------
// Kernel 1: QKV projection.  One wave -> one 16x16 tile of Q, K or V (f16 out).
// M tiles = B*S/16 = 1024, N tiles = 4, matrices = 3  => 12288 waves.
// ---------------------------------------------------------------------------
__global__ __launch_bounds__(256) void qkv_proj_kernel(
    const float* __restrict__ x,
    const float* __restrict__ Wq, const float* __restrict__ Wk,
    const float* __restrict__ Wv,
    _Float16* __restrict__ Qh, _Float16* __restrict__ Kh,
    _Float16* __restrict__ Vh)
{
    const int wid  = (blockIdx.x * blockDim.x + threadIdx.x) >> 5;
    const int lane = threadIdx.x & 31;
    const int hi   = lane >> 4;
    const int l16  = lane & 15;

    const int mat   = wid >> 12;      // 0=Q 1=K 2=V
    const int rem   = wid & 4095;
    const int mtile = rem >> 2;
    const int ntile = rem & 3;

    const float* W = (mat == 0) ? Wq : (mat == 1) ? Wk : Wv;
    _Float16*    O = (mat == 0) ? Qh : (mat == 1) ? Kh : Vh;

    const int arow = mtile * 16 + l16;   // A-frag row for this lane
    const int ncol = ntile * 16 + l16;   // B-frag column for this lane

    F32Frag c;
    #pragma unroll
    for (int r = 0; r < 8; ++r) c.f[r] = 0.0f;

    for (int kb = 0; kb < DM; kb += 32) {
        F16Frag a, b;
        #pragma unroll
        for (int v = 0; v < 8; ++v) {
            const int ka = kb + kmapA(v, hi);
            const float* ap = x + (size_t)arow * DM + ka;
            a.h[2 * v]     = (_Float16)ap[0];
            a.h[2 * v + 1] = (_Float16)ap[1];
            const int kbb = kb + kmapB(v, hi);
            b.h[2 * v]     = (_Float16)W[(size_t)kbb * DH + ncol];
            b.h[2 * v + 1] = (_Float16)W[(size_t)(kbb + 1) * DH + ncol];
        }
        c.v = wmma16(a.v, b.v, c.v);
    }

    // D layout: VGPR r -> row r (lanes 0-15) / r+8 (lanes 16-31); col = lane&15
    #pragma unroll
    for (int r = 0; r < 8; ++r) {
        const int m = mtile * 16 + r + 8 * hi;
        O[(size_t)m * DH + ncol] = (_Float16)c.f[r];
    }
}

// ---------------------------------------------------------------------------
// Kernel 2: causal flash attention.
// One block (8 waves) -> 128 query rows; wave w -> rows [M0+16w, M0+16w+15].
// K/V 32x64 f16 tiles are TDM-DMA'd into LDS (double buffered) by wave 0 and
// shared by all 8 waves; diagonal handled by per-row causal masking.
// ---------------------------------------------------------------------------
__global__ __launch_bounds__(256) void flash_attn_kernel(
    const _Float16* __restrict__ Qh, const _Float16* __restrict__ Kh,
    const _Float16* __restrict__ Vh, float* __restrict__ out)
{
    // [ K0 | K1 | V0 | V1 | P(8 waves) ] = 4KB*4 + 8KB = 24KB
    __shared__ __align__(16) _Float16 smem[2 * 2048 + 2 * 2048 + 8 * 512];

    const unsigned smbase = (unsigned)(size_t)&smem[0];  // LDS byte offset

    const int widb = threadIdx.x >> 5;
    const int lane = threadIdx.x & 31;
    const int hi   = lane >> 4;
    const int l16  = lane & 15;

    const int b  = blockIdx.x >> 5;              // 32 blocks per batch
    const int M0 = (blockIdx.x & 31) * 128;      // block's first query row
    const int m0 = M0 + widb * 16;               // wave's first query row

    const _Float16* Qb = Qh + (size_t)b * SS * DH;
    const _Float16* Kb = Kh + (size_t)b * SS * DH;
    const _Float16* Vb = Vh + (size_t)b * SS * DH;
    _Float16* plds = smem + 8192 + widb * 512;   // wave-private P tile (16x32)

    // Q fragments (16x64 split into two K=32 A-frags), loaded once.
    F16Frag aq0, aq1;
    #pragma unroll
    for (int v = 0; v < 8; ++v) {
        const int k = kmapA(v, hi);
        aq0.u[v] = *(const unsigned*)(Qb + (size_t)(m0 + l16) * DH + k);
        aq1.u[v] = *(const unsigned*)(Qb + (size_t)(m0 + l16) * DH + 32 + k);
    }

    float rowmax[8], rowsum[8];
    F32Frag acc[4];
    #pragma unroll
    for (int r = 0; r < 8; ++r) {
        rowmax[r] = -__builtin_inff();
        rowsum[r] = 0.0f;
        #pragma unroll
        for (int t = 0; t < 4; ++t) acc[t].f[r] = 0.0f;
    }

    const float qk_scale = 0.125f;               // 1/sqrt(64)
    const int jmaxblk = (M0 + 127) >> 5;         // uniform across the block

    // Preload K/V tiles for jb=0 into buffer 0 (wave 0 drives the TDM).
    if (widb == 0) {
        tdm_load_tile(smbase + 0u,    Kb, DH, 32, DH, SS, DH);   // K buf0
        tdm_load_tile(smbase + 8192u, Vb, DH, 32, DH, SS, DH);   // V buf0
    }

    int cur = 0;
    for (int jb = 0; jb <= jmaxblk; ++jb) {
        const int j0 = jb << 5;

        if (widb == 0) {
            if (jb < jmaxblk) {   // kick off next tiles, then wait for current
                const _Float16* nk = Kb + (size_t)(j0 + 32) * DH;
                const _Float16* nv = Vb + (size_t)(j0 + 32) * DH;
                const unsigned nxt = (unsigned)(cur ^ 1) * 4096u;   // byte off
                tdm_load_tile(smbase + nxt,         nk, DH, 32, DH, SS, DH);
                tdm_load_tile(smbase + 8192u + nxt, nv, DH, 32, DH, SS, DH);
                __builtin_amdgcn_s_wait_tensorcnt(2);
            } else {
                __builtin_amdgcn_s_wait_tensorcnt(0);
            }
        }
        __syncthreads();          // buf[cur] visible to all waves

        const _Float16* Kc = smem + cur * 2048;          // 32x64 K tile
        const _Float16* Vc = smem + 4096 + cur * 2048;   // 32x64 V tile

        // ---- scores: S = Q . K^T for columns [j0, j0+32), K from LDS ----
        // B(k,n) = K[(j0+n)*DH + k]; pairs along k are contiguous 32-bit.
        F16Frag bk;
        F32Frag s0, s1;
        const v8f zero = {0.f, 0.f, 0.f, 0.f, 0.f, 0.f, 0.f, 0.f};
        const int col0 = j0 + l16;
        const int col1 = j0 + 16 + l16;

        #pragma unroll
        for (int v = 0; v < 8; ++v)
            bk.u[v] = *(const unsigned*)(Kc + l16 * DH + kmapB(v, hi));
        s0.v = wmma16(aq0.v, bk.v, zero);
        #pragma unroll
        for (int v = 0; v < 8; ++v)
            bk.u[v] = *(const unsigned*)(Kc + l16 * DH + 32 + kmapB(v, hi));
        s0.v = wmma16(aq1.v, bk.v, s0.v);

        #pragma unroll
        for (int v = 0; v < 8; ++v)
            bk.u[v] = *(const unsigned*)(Kc + (16 + l16) * DH + kmapB(v, hi));
        s1.v = wmma16(aq0.v, bk.v, zero);
        #pragma unroll
        for (int v = 0; v < 8; ++v)
            bk.u[v] = *(const unsigned*)(Kc + (16 + l16) * DH + 32 + kmapB(v, hi));
        s1.v = wmma16(aq1.v, bk.v, s1.v);

        // ---- causal mask + online softmax (rows live in VGPR r, half hi) ----
        #pragma unroll
        for (int r = 0; r < 8; ++r) {
            const int rowg = m0 + r + 8 * hi;
            float a0 = s0.f[r] * qk_scale;
            float a1 = s1.f[r] * qk_scale;
            if (col0 > rowg) a0 = -__builtin_inff();
            if (col1 > rowg) a1 = -__builtin_inff();

            const float bmax = red_max16(fmaxf(a0, a1));
            const float newmax = fmaxf(rowmax[r], bmax);
            const float corr = __expf(rowmax[r] - newmax);
            rowmax[r] = newmax;

            const float p0 = __expf(a0 - newmax);
            const float p1 = __expf(a1 - newmax);
            rowsum[r] = rowsum[r] * corr + red_sum16(p0 + p1);

            #pragma unroll
            for (int t = 0; t < 4; ++t) acc[t].f[r] *= corr;

            // spill P row-slice to LDS (D layout -> dense 16x32 f16 tile)
            const int rl = r + 8 * hi;
            plds[rl * 32 + l16]      = (_Float16)p0;
            plds[rl * 32 + 16 + l16] = (_Float16)p1;
        }

        // ---- read P back in A layout, do acc += P . V (V from LDS) ----
        F16Frag pa;
        #pragma unroll
        for (int v = 0; v < 8; ++v)
            pa.u[v] = *(const unsigned*)(plds + l16 * 32 + kmapA(v, hi));

        // B(k,n) = V[(j0+k)*DH + n]; pairs along k are stride-DH 16-bit loads.
        #pragma unroll
        for (int t = 0; t < 4; ++t) {
            F16Frag bv;
            const int nc = t * 16 + l16;
            #pragma unroll
            for (int v = 0; v < 8; ++v) {
                const int k = kmapB(v, hi);
                bv.h[2 * v]     = Vc[k * DH + nc];
                bv.h[2 * v + 1] = Vc[(k + 1) * DH + nc];
            }
            acc[t].v = wmma16(pa.v, bv.v, acc[t].v);
        }

        __syncthreads();          // everyone done with buf[cur] before reuse
        cur ^= 1;
    }

    // ---- finalize: divide by row sums, store f32 ----
    float* ob = out + (size_t)b * SS * DH;
    #pragma unroll
    for (int r = 0; r < 8; ++r) {
        const float inv = 1.0f / rowsum[r];
        const int rowg = m0 + r + 8 * hi;
        #pragma unroll
        for (int t = 0; t < 4; ++t)
            ob[(size_t)rowg * DH + t * 16 + l16] = acc[t].f[r] * inv;
    }
}

// ---------------------------------------------------------------------------
extern "C" void kernel_launch(void* const* d_in, const int* in_sizes, int n_in,
                              void* d_out, int out_size, void* d_ws, size_t ws_size,
                              hipStream_t stream) {
    (void)in_sizes; (void)n_in; (void)out_size; (void)ws_size;

    const float* x  = (const float*)d_in[0];
    const float* Wq = (const float*)d_in[1];
    const float* Wk = (const float*)d_in[2];
    const float* Wv = (const float*)d_in[3];

    _Float16* Qh = (_Float16*)d_ws;                       // 2 MB each
    _Float16* Kh = Qh + (size_t)BB * SS * DH;
    _Float16* Vh = Kh + (size_t)BB * SS * DH;
    float* out = (float*)d_out;

    // 12288 waves: 1024 M-tiles x 4 N-tiles x 3 matrices, 8 waves/block
    qkv_proj_kernel<<<1536, 256, 0, stream>>>(x, Wq, Wk, Wv, Qh, Kh, Vh);
    // 128 blocks: 4 batches x 32 query bands of 128 rows, 8 waves/block
    flash_attn_kernel<<<128, 256, 0, stream>>>(Qh, Kh, Vh, out);
}
